// AttentionDecoder_28965259444467
// MI455X (gfx1250) — matrix operand order repaired and checked by reading, not compiled, not run on hardware
//
#include <hip/hip_runtime.h>
#include <hip/hip_bf16.h>

typedef __attribute__((ext_vector_type(16))) _Float16 v16h;
typedef __attribute__((ext_vector_type(8)))  float    v8f;

#define B_     512
#define T_     512
#define DEC_   128
#define EMB_   512
#define VOCAB_ 30001
#define EXT_   30051   // VOCAB + 50 OOV slots

// ---------------------------------------------------------------------------
// f16-WMMA GEMM, M-blocked:  C[M,N] (+)= A[M,K] @ W[N,K]^T + bias[N]
// fp32 in memory, f16 multiply, f32 accumulate.
// One wave -> 64(M) x 16(N): 4 M-tiles reuse each B fragment (4 WMMAs per
// B load/convert).  Block = 128 threads (4 waves, consecutive N-tiles),
// grid.x = Ntiles/4, grid.y = M/64.  K must be a multiple of 32.
// N tail handled by clamped W-row loads + predicated stores (predication
// only AFTER the WMMAs => EXEC all-1s for every WMMA, as the ISA requires).
// ---------------------------------------------------------------------------
__global__ __launch_bounds__(128) void k_gemm(
    const float* __restrict__ A, const float* __restrict__ W,
    const float* __restrict__ bias, float* __restrict__ C,
    int M, int N, int K, int lda, int ldw, int ldc, int accumulate)
{
  const int lane   = threadIdx.x & 31;
  const int wave   = threadIdx.x >> 5;
  const int lane16 = lane & 15;
  const int half   = lane >> 4;

  const int n_tile = blockIdx.x * 4 + wave;
  const int n0 = n_tile * 16;
  const int m0 = blockIdx.y * 64;
  if (n0 >= N || m0 >= M) return;          // whole-wave uniform exit

  int brow = n0 + lane16;
  const int bvalid = (brow < N);
  if (brow >= N) brow = N - 1;             // clamp: garbage column, store-masked

  const float* __restrict__ Wrow = W + (size_t)brow * ldw;
  const float* __restrict__ Ar[4];
#pragma unroll
  for (int mt = 0; mt < 4; ++mt)
    Ar[mt] = A + (size_t)(m0 + mt * 16 + lane16) * lda;

  v8f acc0 = {}, acc1 = {}, acc2 = {}, acc3 = {};

  for (int kk = 0; kk < K; kk += 32) {
    // prefetch next K-step (global_prefetch_b8)
    if (kk + 32 < K) {
      __builtin_prefetch(Wrow + kk + 32 + half * 16, 0, 1);
#pragma unroll
      for (int mt = 0; mt < 4; ++mt)
        __builtin_prefetch(Ar[mt] + kk + 32 + half * 8, 0, 1);
    }
    // B fragment (= W^T): lane (col=n0+lane16) elements e: K = half*16 + e
    v16h bf;
    const float* bp = Wrow + kk + half * 16;
#pragma unroll
    for (int e = 0; e < 16; ++e) bf[e] = (_Float16)bp[e];

    // 4 M-tiles reuse bf
#pragma unroll
    for (int mt = 0; mt < 4; ++mt) {
      v16h af;
      const float* ap0 = Ar[mt] + kk + half * 16;
      const float* ap1 = Ar[mt] + kk + 8 + half * 16;
#pragma unroll
      for (int e = 0; e < 8; ++e) {
        af[e]     = (_Float16)ap0[e];
        af[e + 8] = (_Float16)ap1[e];
      }
      v8f& acc = (mt == 0) ? acc0 : (mt == 1) ? acc1 : (mt == 2) ? acc2 : acc3;
      acc = __builtin_amdgcn_wmma_f32_16x16x32_f16(
          false, af, false, bf, (short)0, acc, false, false);
    }
  }

  if (bvalid) {
    const int n = n0 + lane16;
    const float bn = bias ? bias[n] : 0.0f;
#pragma unroll
    for (int mt = 0; mt < 4; ++mt) {
      const v8f& acc = (mt == 0) ? acc0 : (mt == 1) ? acc1 : (mt == 2) ? acc2 : acc3;
#pragma unroll
      for (int r = 0; r < 8; ++r) {
        float* cp = C + (size_t)(m0 + mt * 16 + r + 8 * half) * ldc + n;
        float v = acc[r] + bn;
        if (accumulate) v += *cp;
        *cp = v;
      }
    }
  }
}

// ---------------------------------------------------------------------------
// Embedding gather + concat [xe | context_vector]  -> xe(B,512), cat1(B,640)
// ---------------------------------------------------------------------------
__global__ __launch_bounds__(128) void k_embed_cat(
    const float* __restrict__ emb, const float* __restrict__ ctxv,
    const int* __restrict__ x, float* __restrict__ xe, float* __restrict__ cat1)
{
  const int b = blockIdx.x;
  const int xi = x[b];
  const float* __restrict__ erow = emb + (size_t)xi * EMB_;
  for (int j = threadIdx.x; j < EMB_; j += 128) {
    float v = erow[j];
    xe[(size_t)b * EMB_ + j] = v;
    cat1[(size_t)b * 640 + j] = v;
  }
  for (int j = threadIdx.x; j < DEC_; j += 128)
    cat1[(size_t)b * 640 + EMB_ + j] = ctxv[(size_t)b * DEC_ + j];
}

// ---------------------------------------------------------------------------
// LSTM pointwise: gates(B,512), c_prev -> h, c (d_out), s_t=[h,c] (ws)
// ---------------------------------------------------------------------------
__global__ __launch_bounds__(256) void k_lstm(
    const float* __restrict__ gates, const float* __restrict__ c_prev,
    float* __restrict__ h_out, float* __restrict__ c_out, float* __restrict__ s_t)
{
  const int tid = blockIdx.x * blockDim.x + threadIdx.x;
  if (tid >= B_ * DEC_) return;
  const int b = tid / DEC_, j = tid % DEC_;
  const float* g = gates + (size_t)b * 512;
  const float ig = 1.0f / (1.0f + __expf(-g[j]));
  const float fg = 1.0f / (1.0f + __expf(-g[DEC_ + j]));
  const float gg = tanhf(g[2 * DEC_ + j]);
  const float og = 1.0f / (1.0f + __expf(-g[3 * DEC_ + j]));
  const float c = fg * c_prev[tid] + ig * gg;
  const float h = og * tanhf(c);
  h_out[tid] = h;
  c_out[tid] = c;
  s_t[(size_t)b * 256 + j] = h;
  s_t[(size_t)b * 256 + DEC_ + j] = c;
}

// ---------------------------------------------------------------------------
// Fused coverage-attention per batch row (one block per b, 8 waves):
//   enc_f = h_i[b] @ Wh^T (WMMA, never materialized)
//   e[t]  = sum_d V_w[d]*tanh(enc_f + dec_f[d] + cov[t]*Wc[d])
//   a = softmax(e); coverage_new = cov + a; ctx[d] = sum_t a[t]*h_i[b,t,d]
// ---------------------------------------------------------------------------
__global__ __launch_bounds__(256) void k_attention(
    const float* __restrict__ h_i, const float* __restrict__ Wh_w,
    const float* __restrict__ dec_f, const float* __restrict__ coverage,
    const float* __restrict__ Wc_w, const float* __restrict__ V_w,
    float* __restrict__ a_t, float* __restrict__ coverage_new,
    float* __restrict__ ctx)
{
  __shared__ float e_buf[T_];
  __shared__ float cov_s[T_];
  __shared__ float dec_s[DEC_];
  __shared__ float red[256];
  __shared__ float scal[2];

  const int b   = blockIdx.x;
  const int tid = threadIdx.x;
  const int wave = tid >> 5;
  const int lane16 = tid & 15;
  const int half = (tid >> 4) & 1;

  for (int t = tid; t < T_; t += 256) cov_s[t] = coverage[(size_t)b * T_ + t];
  if (tid < DEC_) dec_s[tid] = dec_f[(size_t)b * DEC_ + tid];
  __syncthreads();

  // ---- e_t via WMMA, 8 waves x 4 t-tiles of 16 ----
  for (int tt = wave; tt < T_ / 16; tt += 8) {
    const int t0 = tt * 16;
    const float* __restrict__ Arow =
        h_i + ((size_t)b * T_ + t0 + lane16) * DEC_;
    float part[8];
#pragma unroll
    for (int r = 0; r < 8; ++r) part[r] = 0.0f;

    for (int nt = 0; nt < DEC_ / 16; ++nt) {       // output-d tiles
      const int n0 = nt * 16;
      const float* __restrict__ Wrow = Wh_w + (size_t)(n0 + lane16) * DEC_;
      v8f acc = {};
#pragma unroll
      for (int kk = 0; kk < DEC_; kk += 32) {
        v16h af, bf;
        const float* ap0 = Arow + kk + half * 16;
        const float* ap1 = Arow + kk + 8 + half * 16;
#pragma unroll
        for (int e = 0; e < 8; ++e) { af[e] = (_Float16)ap0[e]; af[e+8] = (_Float16)ap1[e]; }
        const float* bp = Wrow + kk + half * 16;
#pragma unroll
        for (int e = 0; e < 16; ++e) bf[e] = (_Float16)bp[e];
        acc = __builtin_amdgcn_wmma_f32_16x16x32_f16(
            false, af, false, bf, (short)0, acc, false, false);
      }
      // acc[r] = enc_f[t0 + r + 8*half][d],  d = n0 + lane16
      const int d = n0 + lane16;
      const float vw = V_w[d], wc = Wc_w[d], df = dec_s[d];
#pragma unroll
      for (int r = 0; r < 8; ++r) {
        const int t = t0 + r + 8 * half;
        part[r] += vw * tanhf(acc[r] + df + cov_s[t] * wc);
      }
    }
    // reduce over d: sum across the 16 lanes of each half-wave
#pragma unroll
    for (int off = 8; off > 0; off >>= 1) {
#pragma unroll
      for (int r = 0; r < 8; ++r) part[r] += __shfl_xor(part[r], off, 16);
    }
    if (lane16 == 0) {
#pragma unroll
      for (int r = 0; r < 8; ++r) e_buf[t0 + r + 8 * half] = part[r];
    }
  }
  __syncthreads();

  // ---- softmax over T=512 with 256 threads ----
  const float v0 = e_buf[tid], v1 = e_buf[tid + 256];
  red[tid] = fmaxf(v0, v1);
  __syncthreads();
  for (int o = 128; o > 0; o >>= 1) {
    if (tid < o) red[tid] = fmaxf(red[tid], red[tid + o]);
    __syncthreads();
  }
  if (tid == 0) scal[0] = red[0];
  __syncthreads();
  const float mx = scal[0];
  const float e0 = __expf(v0 - mx), e1 = __expf(v1 - mx);
  red[tid] = e0 + e1;
  __syncthreads();
  for (int o = 128; o > 0; o >>= 1) {
    if (tid < o) red[tid] += red[tid + o];
    __syncthreads();
  }
  if (tid == 0) scal[1] = 1.0f / red[0];
  __syncthreads();
  const float inv = scal[1];
  const float a0 = e0 * inv, a1 = e1 * inv;
  a_t[(size_t)b * T_ + tid]       = a0;
  a_t[(size_t)b * T_ + tid + 256] = a1;
  coverage_new[(size_t)b * T_ + tid]       = cov_s[tid] + a0;
  coverage_new[(size_t)b * T_ + tid + 256] = cov_s[tid + 256] + a1;
  e_buf[tid] = a0;
  e_buf[tid + 256] = a1;
  __syncthreads();

  // ---- ctx[d] = sum_t a[t]*h_i[b,t,d]; 256 thr = (d, t-segment) split ----
  {
    const int d = tid & 127, seg = tid >> 7;      // seg 0: t<256, seg 1: t>=256
    float s = 0.0f;
    const float* __restrict__ hb =
        h_i + (size_t)b * T_ * DEC_ + (size_t)seg * 256 * DEC_ + d;
    const float* __restrict__ ab = e_buf + seg * 256;
    for (int t = 0; t < 256; ++t) s += ab[t] * hb[(size_t)t * DEC_];
    red[tid] = s;
  }
  __syncthreads();
  if (tid < DEC_) ctx[(size_t)b * DEC_ + tid] = red[tid] + red[tid + 128];
}

// ---------------------------------------------------------------------------
// Build V1 input [h | ctx] (B,256)
// ---------------------------------------------------------------------------
__global__ __launch_bounds__(256) void k_v1in(
    const float* __restrict__ h, const float* __restrict__ ctx,
    float* __restrict__ v1in)
{
  const int tid = blockIdx.x * blockDim.x + threadIdx.x;
  if (tid >= B_ * 256) return;
  const int b = tid >> 8, j = tid & 255;
  v1in[tid] = (j < DEC_) ? h[(size_t)b * DEC_ + j]
                         : ctx[(size_t)b * DEC_ + (j - DEC_)];
}

// ---------------------------------------------------------------------------
// p_gen[b] = sigmoid( dot([ctx, s_t, xe], W2_w) + W2_b )   (896-wide dot)
// ---------------------------------------------------------------------------
__global__ __launch_bounds__(128) void k_pgen(
    const float* __restrict__ ctx, const float* __restrict__ s_t,
    const float* __restrict__ xe, const float* __restrict__ W2_w,
    const float* __restrict__ W2_b, float* __restrict__ p_gen)
{
  __shared__ float red[128];
  const int b = blockIdx.x, tid = threadIdx.x;
  float s = 0.0f;
  for (int k = tid; k < 896; k += 128) {
    float v;
    if (k < 128)       v = ctx[(size_t)b * DEC_ + k];
    else if (k < 384)  v = s_t[(size_t)b * 256 + (k - 128)];
    else               v = xe[(size_t)b * EMB_ + (k - 384)];
    s += v * W2_w[k];
  }
  red[tid] = s;
  __syncthreads();
  for (int o = 64; o > 0; o >>= 1) {
    if (tid < o) red[tid] += red[tid + o];
    __syncthreads();
  }
  if (tid == 0) p_gen[b] = 1.0f / (1.0f + __expf(-(red[0] + W2_b[0])));
}

// ---------------------------------------------------------------------------
// Final head per row b: online softmax over 30001 logits (in-place in d_out),
// scale by p_gen, zero 50 OOV slots, write attn_dist, scatter-add via atomics.
// ---------------------------------------------------------------------------
__global__ __launch_bounds__(256) void k_finalize(
    float* __restrict__ final_dist, float* __restrict__ attn_out,
    const float* __restrict__ a_t, const float* __restrict__ p_gen,
    const int* __restrict__ code_ext)
{
  __shared__ float sm[256], ss[256];
  const int b = blockIdx.x, tid = threadIdx.x;
  float* __restrict__ row = final_dist + (size_t)b * EXT_;
  const float pg = p_gen[b];
  const float om = 1.0f - pg;

  for (int t = tid; t < T_; t += 256)
    attn_out[(size_t)b * T_ + t] = om * a_t[(size_t)b * T_ + t];

  // online max/sum over logits
  float m = -INFINITY, s = 0.0f;
  for (int k = tid; k < VOCAB_; k += 256) {
    const float v = row[k];
    const float nm = fmaxf(m, v);
    s = s * __expf(m - nm) + __expf(v - nm);
    m = nm;
  }
  sm[tid] = m; ss[tid] = s;
  __syncthreads();
  for (int o = 128; o > 0; o >>= 1) {
    if (tid < o) {
      const float m2 = sm[tid + o], s2 = ss[tid + o];
      const float nm = fmaxf(sm[tid], m2);
      ss[tid] = ss[tid] * __expf(sm[tid] - nm) + s2 * __expf(m2 - nm);
      sm[tid] = nm;
    }
    __syncthreads();
  }
  const float M = sm[0];
  const float scale = pg / ss[0];

  for (int k = tid; k < VOCAB_; k += 256) row[k] = __expf(row[k] - M) * scale;
  for (int k = VOCAB_ + tid; k < EXT_; k += 256) row[k] = 0.0f;
  __syncthreads();

  for (int t = tid; t < T_; t += 256) {
    const int ci = code_ext[(size_t)b * T_ + t];
    atomicAdd(&row[ci], om * a_t[(size_t)b * T_ + t]);
  }
}

// ---------------------------------------------------------------------------
extern "C" void kernel_launch(void* const* d_in, const int* in_sizes, int n_in,
                              void* d_out, int out_size, void* d_ws, size_t ws_size,
                              hipStream_t stream) {
  (void)in_sizes; (void)n_in; (void)out_size; (void)ws_size;
  const float* h_i    = (const float*)d_in[0];
  const float* h_prev = (const float*)d_in[1];
  const float* c_prev = (const float*)d_in[2];
  const float* ctxv   = (const float*)d_in[3];
  const float* cover  = (const float*)d_in[4];
  const float* emb    = (const float*)d_in[5];
  const float* W_ih   = (const float*)d_in[6];
  const float* W_hh   = (const float*)d_in[7];
  const float* b_ih   = (const float*)d_in[8];
  const float* b_hh   = (const float*)d_in[9];
  const float* W1_w   = (const float*)d_in[10];
  const float* W1_b   = (const float*)d_in[11];
  const float* W2_w   = (const float*)d_in[12];
  const float* W2_b   = (const float*)d_in[13];
  const float* V1_w   = (const float*)d_in[14];
  const float* V1_b   = (const float*)d_in[15];
  const float* V2_w   = (const float*)d_in[16];
  const float* V2_b   = (const float*)d_in[17];
  const float* Wh_w   = (const float*)d_in[18];
  const float* Ws_w   = (const float*)d_in[19];
  const float* Ws_b   = (const float*)d_in[20];
  const float* Wc_w   = (const float*)d_in[21];
  const float* V_w    = (const float*)d_in[22];
  const int*   x      = (const int*)d_in[23];
  const int*   code   = (const int*)d_in[24];

  // d_out layout (floats): final | attn | h | c | coverage_new | ctx
  float* out        = (float*)d_out;
  float* o_final    = out;
  float* o_attn     = o_final + (size_t)B_ * EXT_;
  float* o_h        = o_attn  + (size_t)B_ * T_;
  float* o_c        = o_h     + (size_t)B_ * DEC_;
  float* o_covnew   = o_c     + (size_t)B_ * DEC_;
  float* o_ctx      = o_covnew + (size_t)B_ * T_;

  // workspace layout (floats)
  float* ws     = (float*)d_ws;
  float* w_xe   = ws;                          // B*512
  float* w_cat1 = w_xe   + (size_t)B_ * EMB_;  // B*640
  float* w_x1   = w_cat1 + (size_t)B_ * 640;   // B*512
  float* w_gate = w_x1   + (size_t)B_ * 512;   // B*512
  float* w_st   = w_gate + (size_t)B_ * 512;   // B*256
  float* w_decf = w_st   + (size_t)B_ * 256;   // B*128
  float* w_at   = w_decf + (size_t)B_ * DEC_;  // B*512
  float* w_v1in = w_at   + (size_t)B_ * T_;    // B*256
  float* w_pv   = w_v1in + (size_t)B_ * 256;   // B*384
  float* w_pg   = w_pv   + (size_t)B_ * 384;   // B

  const dim3 blk128(128), blk256(256);

  // 1. embedding gather + concat
  k_embed_cat<<<B_, blk128, 0, stream>>>(emb, ctxv, x, w_xe, w_cat1);

  // 2. x1 = cat1 @ W1^T + W1_b          (M=512,N=512,K=640)
  k_gemm<<<dim3(512/16/4, B_/64), blk128, 0, stream>>>(
      w_cat1, W1_w, W1_b, w_x1, B_, 512, 640, 640, 640, 512, 0);

  // 3. gates = x1 @ W_ih^T + b_ih       (M=512,N=512,K=512)
  k_gemm<<<dim3(512/16/4, B_/64), blk128, 0, stream>>>(
      w_x1, W_ih, b_ih, w_gate, B_, 512, 512, 512, 512, 512, 0);

  // 4. gates += h_prev @ W_hh^T + b_hh  (K=128, accumulate)
  k_gemm<<<dim3(512/16/4, B_/64), blk128, 0, stream>>>(
      h_prev, W_hh, b_hh, w_gate, B_, 512, 128, 128, 128, 512, 1);

  // 5. LSTM pointwise -> h, c, s_t
  k_lstm<<<(B_ * DEC_) / 256, blk256, 0, stream>>>(w_gate, c_prev, o_h, o_c, w_st);

  // 6. dec_f = s_t @ Ws^T + Ws_b        (M=512,N=128,K=256)
  k_gemm<<<dim3(128/16/4, B_/64), blk128, 0, stream>>>(
      w_st, Ws_w, Ws_b, w_decf, B_, 128, 256, 256, 256, 128, 0);

  // 7. fused attention (WMMA enc_f + softmax + coverage + ctx)
  k_attention<<<B_, blk256, 0, stream>>>(
      h_i, Wh_w, w_decf, cover, Wc_w, V_w, w_at, o_covnew, o_ctx);

  // 8. V1 input + p_gen (independent)
  k_v1in<<<(B_ * 256) / 256, blk256, 0, stream>>>(o_h, o_ctx, w_v1in);
  k_pgen<<<B_, blk128, 0, stream>>>(o_ctx, w_st, w_xe, W2_w, W2_b, w_pg);

  // 9. pv_pre = v1in @ V1^T + V1_b      (M=512,N=384,K=256)
  k_gemm<<<dim3(384/16/4, B_/64), blk128, 0, stream>>>(
      w_v1in, V1_w, V1_b, w_pv, B_, 384, 256, 256, 256, 384, 0);

  // 10. logits = pv_pre @ V2^T + V2_b   (M=512,N=30001,K=384) -> d_out rows
  //     Ntiles = ceil(30001/16) = 1876, grid.x = 1876/4 = 469
  k_gemm<<<dim3(469, B_/64), blk128, 0, stream>>>(
      w_pv, V2_w, V2_b, o_final, B_, VOCAB_, 384, 384, 384, EXT_, 0);

  // 11. softmax * p_gen, OOV zeros, attn_dist, scatter-add
  k_finalize<<<B_, blk256, 0, stream>>>(o_final, o_attn, w_at, w_pg, code);
}
